// LSTMPredictor_45638322487551
// MI455X (gfx1250) — matrix operand order repaired and checked by reading, not compiled, not run on hardware
//
#include <hip/hip_runtime.h>
#include <hip/hip_bf16.h>

typedef __attribute__((ext_vector_type(16))) _Float16 v16h;
typedef __attribute__((ext_vector_type(8)))  float    v8f;
typedef __attribute__((ext_vector_type(4)))  float    f4;
typedef __attribute__((ext_vector_type(4)))  _Float16 h4;

#define T_STEPS 25
#define STR 168               // stage row stride in halves (>=160, mult of 8)

// fragment offsets in d_ws (units of 512 halves = one 32x16 f16 B fragment)
#define OFF0 0                // LSTM1 gates: KC=1, NT=4
#define OFF1 4                // W1:          KC=2, NT=4
#define OFF2 12               // W2:          KC=2, NT=5
#define OFF3 22               // LSTM2 gates: KC=5, NT=20 (gates padded 72->80)
#define OFF4 122              // LSTM3 gates: KC=4, NT=8
#define OFF5 154              // W3:          KC=1, NT=1
#define OFF6 155              // W4:          KC=1, NT=1
#define NFRAG 156

#define LOG2E 1.4426950408889634f

// DS ops from one wave execute in order (ISA 7.3): only a compiler barrier is
// needed between cross-lane LDS store->load phases; hardware waits for the
// loads' own results are inserted by the compiler.
__device__ __forceinline__ void compiler_fence() {
  asm volatile("" ::: "memory");
}

__device__ __forceinline__ v8f wmma_f16(v16h a, v16h b, v8f c) {
  return __builtin_amdgcn_wmma_f32_16x16x32_f16(false, a, false, b,
                                                (short)0, c, false, false);
}

// zero accumulator: lowers to WMMA SRC2 = inline 0
__device__ __forceinline__ v8f zacc() {
  v8f v;
#pragma unroll
  for (int e = 0; e < 8; ++e) v[e] = 0.f;
  return v;
}

// branch-free, hardware-native activations (v_exp_f32 computes 2^x)
__device__ __forceinline__ float sigm(float x) {
  return __builtin_amdgcn_rcpf(1.0f + __builtin_amdgcn_exp2f(-LOG2E * x));
}
__device__ __forceinline__ float tanh_(float x) {
  // tanh(x) = 1 - 2/(exp(2x)+1); saturates correctly for |x| large
  return 1.0f - 2.0f * __builtin_amdgcn_rcpf(1.0f + __builtin_amdgcn_exp2f(2.0f * LOG2E * x));
}

// B fragment: lane holds column N=lane&15, 16 contiguous K halves
__device__ __forceinline__ v16h load_b(const _Float16* __restrict__ wsB, int frag, int lane) {
  return *(const v16h*)(wsB + ((size_t)frag << 9) + (lane << 4));
}

// A fragment from row-major f16 stage tile: lane holds row M=lane&15, 16 contiguous K halves (matches B)
__device__ __forceinline__ v16h load_a(const _Float16* stg, int lane, int kc) {
  const _Float16* p = stg + (lane & 15) * STR + kc * 32 + ((lane & 16) ? 16 : 0);
  v16h a;
#pragma unroll
  for (int i = 0; i < 8; ++i) a[i] = p[i];
#pragma unroll
  for (int i = 0; i < 8; ++i) a[8 + i] = p[8 + i];
  return a;
}

// C fragment (f32) -> row-major f16 stage tile, column-guarded
__device__ __forceinline__ void stage_frag(_Float16* stg, const v8f& v, int lane,
                                           int dstColBase, int nvalid) {
  int n  = lane & 15;
  int mo = (lane & 16) >> 1;        // 0 or 8
  if (n < nvalid) {
    int col = dstColBase + n;
#pragma unroll
    for (int e = 0; e < 8; ++e) stg[(mo + e) * STR + col] = (_Float16)v[e];
  }
}

// zero an aligned column range [C0, C0+NCOLS) with b64 stores; NCOLS/4 pow2
template<int C0, int NCOLS>
__device__ __forceinline__ void zero_cols4(_Float16* stg, int lane) {
  constexpr int N4 = NCOLS / 4;
  constexpr int TOTAL = N4 * 16;
  h4 z;
#pragma unroll
  for (int j = 0; j < 4; ++j) z[j] = (_Float16)0.f;
#pragma unroll
  for (int f = lane; f < TOTAL; f += 32) {
    int r = f / N4, q = f % N4;
    *(h4*)(stg + r * STR + C0 + q * 4) = z;
  }
}

// gates come in WITHOUT bias; bias scalars are folded into the activations
__device__ __forceinline__ void lstm_update(v8f& h, v8f& c,
                                            const v8f& gi, const v8f& gf,
                                            const v8f& gg, const v8f& go,
                                            float bi, float bf, float bg, float bo) {
#pragma unroll
  for (int e = 0; e < 8; ++e) {
    float cn = sigm(gf[e] + bf) * c[e] + sigm(gi[e] + bi) * tanh_(gg[e] + bg);
    c[e] = cn;
    h[e] = sigm(go[e] + bo) * tanh_(cn);
  }
}

// ---------------------------------------------------------------------------
// Weight prep: fp32 weights -> f16 B-fragment layout in d_ws
// ---------------------------------------------------------------------------
__global__ __launch_bounds__(512) void prep_weights(
    const float* __restrict__ Wih1, const float* __restrict__ Whh1,
    const float* __restrict__ Wih2, const float* __restrict__ Whh2,
    const float* __restrict__ Wih3, const float* __restrict__ Whh3,
    const float* __restrict__ W1,   const float* __restrict__ W2,
    const float* __restrict__ W3,   const float* __restrict__ W4,
    _Float16* __restrict__ wsB)
{
  int frag = blockIdx.x;
  int lane = threadIdx.x >> 4;
  int hh   = threadIdx.x & 15;
  int n15  = lane & 15;
  int kb   = (lane & 16) ? 16 : 0;
  float val = 0.f;

  if (frag < OFF1) {                       // LSTM1: A=[x(2) | h1(16)], N=64
    int nt = frag - OFF0;
    int K = kb + hh, N = nt * 16 + n15;
    if (K < 2)       val = Wih1[N * 2 + K];
    else if (K < 18) val = Whh1[N * 16 + (K - 2)];
  } else if (frag < OFF2) {                // W1: K=36->64, N=64
    int l = frag - OFF1; int kc = l >> 2, nt = l & 3;
    int K = kc * 32 + kb + hh, N = nt * 16 + n15;
    if (K < 36) val = W1[N * 36 + K];
  } else if (frag < OFF3) {                // W2: K=64, N=72->80
    int l = frag - OFF2; int kc = l / 5, nt = l % 5;
    int K = kc * 32 + kb + hh, N = nt * 16 + n15;
    if (N < 72 && K < 64) val = W2[N * 64 + K];
  } else if (frag < OFF4) {                // LSTM2: A=[l2(72)|h2(72)]->160, gates 72->80 each
    int l = frag - OFF3; int kc = l / 20, nt = l % 20;
    int K = kc * 32 + kb + hh, N = nt * 16 + n15;
    int gate = N / 80, j = N % 80;
    if (j < 72 && K < 144) {
      int row = gate * 72 + j;
      val = (K < 72) ? Wih2[row * 72 + K] : Whh2[row * 72 + (K - 72)];
    }
  } else if (frag < OFF5) {                // LSTM3: A=[h1(16)|h2(72)|h3(32)]->128, N=128
    int l = frag - OFF4; int kc = l >> 3, nt = l & 7;
    int K = kc * 32 + kb + hh, N = nt * 16 + n15;
    if (K < 88)       val = Wih3[N * 88 + K];
    else if (K < 120) val = Whh3[N * 32 + (K - 88)];
  } else if (frag == OFF5) {               // W3: K=32, N=16
    int K = kb + hh, N = n15;
    val = W3[N * 32 + K];
  } else {                                 // W4: K=16->32, N=2->16
    int K = kb + hh, N = n15;
    if (K < 16 && N < 2) val = W4[N * 16 + K];
  }
  wsB[(size_t)frag * 512 + threadIdx.x] = (_Float16)val;
}

// ---------------------------------------------------------------------------
// Fused 25-step LSTM stack: one wave = one 16-row batch tile, state in VGPRs
// ---------------------------------------------------------------------------
__global__ __launch_bounds__(256, 1) void lstm_fused(
    const float* __restrict__ x,    const float* __restrict__ y,
    const float* __restrict__ bih1, const float* __restrict__ bhh1,
    const float* __restrict__ bih2, const float* __restrict__ bhh2,
    const float* __restrict__ bih3, const float* __restrict__ bhh3,
    const float* __restrict__ b1,   const float* __restrict__ b2,
    const float* __restrict__ b3,   const float* __restrict__ b4,
    const _Float16* __restrict__ wsB,
    float* __restrict__ out)
{
  __shared__ _Float16 stageAll[8 * 16 * STR];

  const int lane = threadIdx.x & 31;
  const int wave = threadIdx.x >> 5;
  const int n15  = lane & 15;
  const int mo   = (lane & 16) >> 1;     // 0 or 8 (C-fragment row offset)
  const bool hiHalf = (lane & 16) != 0;
  _Float16* stg  = stageAll + wave * (16 * STR);
  const int rowBase = blockIdx.x * 128 + wave * 16;

  for (int i = lane; i < 16 * STR; i += 32) stg[i] = (_Float16)0.f;

  // per-step input base pointers (advance by t inside the loop)
  const float* xp = x + (size_t)(rowBase + (lane >> 1)) * (T_STEPS * 2) + (lane & 1);
  const int   ycb = hiHalf ? 20 : 0;     // half-wave column split of the 36-wide y row
  const int   ytailsrc = hiHalf ? 0 : 16;   // dummy-safe tail load offset
  const int   ytaildst = hiHalf ? 160 : 16; // hi half writes to never-read pad cols
  const float* yp = y + (size_t)(rowBase + n15) * (T_STEPS * 36) + ycb;
  float* outp = out + (size_t)(rowBase + mo) * (2 * T_STEPS) + n15;

  // hoisted per-lane scalar biases (column = n15 within each 16-wide tile)
  float bG0[4], bG1v[4], bG2v[5], bG3v[20], bG4v[8], bG5v, bG6v;
#pragma unroll
  for (int nt = 0; nt < 4; ++nt) bG0[nt]  = bih1[nt * 16 + n15] + bhh1[nt * 16 + n15];
#pragma unroll
  for (int nt = 0; nt < 4; ++nt) bG1v[nt] = b1[nt * 16 + n15];
#pragma unroll
  for (int nt = 0; nt < 5; ++nt) { int n = nt * 16 + n15; bG2v[nt] = (n < 72) ? b2[n] : 0.f; }
#pragma unroll
  for (int nt = 0; nt < 20; ++nt) {
    int gate = nt / 5, j = (nt % 5) * 16 + n15;
    bG3v[nt] = (j < 72) ? (bih2[gate * 72 + j] + bhh2[gate * 72 + j]) : 0.f;
  }
#pragma unroll
  for (int nt = 0; nt < 8; ++nt) bG4v[nt] = bih3[nt * 16 + n15] + bhh3[nt * 16 + n15];
  bG5v = b3[n15];
  bG6v = (n15 < 2) ? b4[n15] : 0.f;

  // recurrent state as C fragments (f32)
  v8f h1 = zacc(), c1 = zacc();
  v8f h2[5], c2[5], h3[2], c3[2];
#pragma unroll
  for (int i = 0; i < 5; ++i) { h2[i] = zacc(); c2[i] = zacc(); }
#pragma unroll
  for (int i = 0; i < 2; ++i) { h3[i] = zacc(); c3[i] = zacc(); }

  compiler_fence();

  for (int t = 0; t < T_STEPS; ++t) {
    // ---- prefetch this step's inputs (hidden behind G0 compute) ---------
    float xv = xp[t * 2];
    const float* yrow = yp + t * 36;
    f4 yv[5];
#pragma unroll
    for (int q = 0; q < 4; ++q) yv[q] = *(const f4*)(yrow + q * 4);
    yv[4] = *(const f4*)(yrow + ytailsrc);   // unconditional; safe in-bounds

    // ---- G0: LSTM1 gates = [x_t | h1] @ B0,  K=18->32, N=64 -------------
    zero_cols4<0, 32>(stg, lane);
    compiler_fence();
    stg[(lane >> 1) * STR + (lane & 1)] = (_Float16)xv;
    stage_frag(stg, h1, lane, 2, 16);
    compiler_fence();
    {
      v16h a0 = load_a(stg, lane, 0);
      v16h bi = load_b(wsB, OFF0 + 0, lane), bf = load_b(wsB, OFF0 + 1, lane);
      v16h bg = load_b(wsB, OFF0 + 2, lane), bo = load_b(wsB, OFF0 + 3, lane);
      v8f gi = wmma_f16(a0, bi, zacc());
      v8f gf = wmma_f16(a0, bf, zacc());
      v8f gg = wmma_f16(a0, bg, zacc());
      v8f go = wmma_f16(a0, bo, zacc());
      lstm_update(h1, c1, gi, gf, gg, go, bG0[0], bG0[1], bG0[2], bG0[3]);
    }
    compiler_fence();

    // ---- G1: z1 = relu(y_t @ W1.T + b1),  K=36->64, N=64 ----------------
    zero_cols4<32, 32>(stg, lane);
    compiler_fence();
#pragma unroll
    for (int q = 0; q < 4; ++q) {
      h4 hv;
#pragma unroll
      for (int j = 0; j < 4; ++j) hv[j] = (_Float16)yv[q][j];
      *(h4*)(stg + n15 * STR + ycb + q * 4) = hv;
    }
    {
      // branch-free tail: upper half-wave writes to unused pad cols 160..163
      h4 hv;
#pragma unroll
      for (int j = 0; j < 4; ++j) hv[j] = (_Float16)yv[4][j];
      *(h4*)(stg + n15 * STR + ytaildst) = hv;
    }
    compiler_fence();
    v8f z1[4];
    {
      v16h a0 = load_a(stg, lane, 0), a1 = load_a(stg, lane, 1);
#pragma unroll
      for (int nt = 0; nt < 4; ++nt) {
        v16h b0 = load_b(wsB, OFF1 + 0 * 4 + nt, lane);
        v16h b1f = load_b(wsB, OFF1 + 1 * 4 + nt, lane);
        v8f acc = wmma_f16(a0, b0, zacc());
        acc = wmma_f16(a1, b1f, acc);
#pragma unroll
        for (int e = 0; e < 8; ++e) acc[e] = fmaxf(acc[e] + bG1v[nt], 0.f);
        z1[nt] = acc;
      }
    }
    compiler_fence();

    // ---- G2: l2 = z1 @ W2.T + b2,  K=64, N=72->80 -----------------------
#pragma unroll
    for (int nt = 0; nt < 4; ++nt) stage_frag(stg, z1[nt], lane, nt * 16, 16);
    compiler_fence();
    v8f l2[5];
    {
      v16h a0 = load_a(stg, lane, 0), a1 = load_a(stg, lane, 1);
#pragma unroll
      for (int nt = 0; nt < 5; ++nt) {
        v16h b0 = load_b(wsB, OFF2 + 0 * 5 + nt, lane);
        v16h b1f = load_b(wsB, OFF2 + 1 * 5 + nt, lane);
        v8f acc = wmma_f16(a0, b0, zacc());
        acc = wmma_f16(a1, b1f, acc);
#pragma unroll
        for (int e = 0; e < 8; ++e) acc[e] += bG2v[nt];
        l2[nt] = acc;
      }
    }
    compiler_fence();

    // ---- G3: LSTM2 gates = [l2 | h2_prev] @ B3,  K=144->160, N=320 ------
    zero_cols4<144, 16>(stg, lane);
    compiler_fence();
#pragma unroll
    for (int nt = 0; nt < 5; ++nt) {
      int nv = 72 - nt * 16; nv = nv > 16 ? 16 : nv;
      stage_frag(stg, l2[nt], lane, nt * 16, nv);          // cols 0..71
      stage_frag(stg, h2[nt], lane, 72 + nt * 16, nv);     // cols 72..143
    }
    compiler_fence();
    {
      v16h a3[5];
#pragma unroll
      for (int kc = 0; kc < 5; ++kc) a3[kc] = load_a(stg, lane, kc);
#pragma unroll
      for (int nt5 = 0; nt5 < 5; ++nt5) {
        v8f gi = zacc(), gf = zacc(), gg = zacc(), go = zacc();
#pragma unroll
        for (int kc = 0; kc < 5; ++kc) {
          int base = OFF3 + kc * 20;
          // batch the 4 B-fragment loads so they overlap the WMMA chain
          v16h bi = load_b(wsB, base + nt5, lane);
          v16h bf = load_b(wsB, base + 5 + nt5, lane);
          v16h bg = load_b(wsB, base + 10 + nt5, lane);
          v16h bo = load_b(wsB, base + 15 + nt5, lane);
          gi = wmma_f16(a3[kc], bi, gi);
          gf = wmma_f16(a3[kc], bf, gf);
          gg = wmma_f16(a3[kc], bg, gg);
          go = wmma_f16(a3[kc], bo, go);
        }
        lstm_update(h2[nt5], c2[nt5], gi, gf, gg, go,
                    bG3v[nt5], bG3v[5 + nt5], bG3v[10 + nt5], bG3v[15 + nt5]);
      }
    }
    compiler_fence();

    // ---- G4: LSTM3 gates = [h1 | h2 | h3_prev] @ B4,  K=120->128, N=128 -
    zero_cols4<120, 8>(stg, lane);
    compiler_fence();
    stage_frag(stg, h1, lane, 0, 16);
#pragma unroll
    for (int nt = 0; nt < 5; ++nt) {
      int nv = 72 - nt * 16; nv = nv > 16 ? 16 : nv;
      stage_frag(stg, h2[nt], lane, 16 + nt * 16, nv);
    }
    stage_frag(stg, h3[0], lane, 88, 16);
    stage_frag(stg, h3[1], lane, 104, 16);
    compiler_fence();
    {
      v16h a4[4];
#pragma unroll
      for (int kc = 0; kc < 4; ++kc) a4[kc] = load_a(stg, lane, kc);
#pragma unroll
      for (int nt2 = 0; nt2 < 2; ++nt2) {
        v8f gi = zacc(), gf = zacc(), gg = zacc(), go = zacc();
#pragma unroll
        for (int kc = 0; kc < 4; ++kc) {
          int base = OFF4 + kc * 8;
          v16h bi = load_b(wsB, base + nt2, lane);
          v16h bf = load_b(wsB, base + 2 + nt2, lane);
          v16h bg = load_b(wsB, base + 4 + nt2, lane);
          v16h bo = load_b(wsB, base + 6 + nt2, lane);
          gi = wmma_f16(a4[kc], bi, gi);
          gf = wmma_f16(a4[kc], bf, gf);
          gg = wmma_f16(a4[kc], bg, gg);
          go = wmma_f16(a4[kc], bo, go);
        }
        lstm_update(h3[nt2], c3[nt2], gi, gf, gg, go,
                    bG4v[nt2], bG4v[2 + nt2], bG4v[4 + nt2], bG4v[6 + nt2]);
      }
    }
    compiler_fence();

    // ---- G5: z3 = relu(h3 @ W3.T + b3),  K=32, N=16 ---------------------
    stage_frag(stg, h3[0], lane, 0, 16);
    stage_frag(stg, h3[1], lane, 16, 16);
    compiler_fence();
    v8f z3;
    {
      v16h a5 = load_a(stg, lane, 0);
      v16h b5 = load_b(wsB, OFF5, lane);
      v8f acc = wmma_f16(a5, b5, zacc());
#pragma unroll
      for (int e = 0; e < 8; ++e) acc[e] = fmaxf(acc[e] + bG5v, 0.f);
      z3 = acc;
    }
    compiler_fence();

    // ---- G6: out_t = z3 @ W4.T + b4,  K=16->32, N=2->16 -----------------
    zero_cols4<16, 16>(stg, lane);
    compiler_fence();
    stage_frag(stg, z3, lane, 0, 16);
    compiler_fence();
    {
      v16h a6 = load_a(stg, lane, 0);
      v16h b6 = load_b(wsB, OFF6, lane);
      v8f o = wmma_f16(a6, b6, zacc());
      if (n15 < 2) {
#pragma unroll
        for (int e = 0; e < 8; ++e)
          outp[(size_t)e * (2 * T_STEPS) + t * 2] = o[e] + bG6v;
      }
    }
    compiler_fence();
  }
}

extern "C" void kernel_launch(void* const* d_in, const int* in_sizes, int n_in,
                              void* d_out, int out_size, void* d_ws, size_t ws_size,
                              hipStream_t stream) {
  const float* x    = (const float*)d_in[0];
  const float* y    = (const float*)d_in[1];
  const float* Wih1 = (const float*)d_in[2];
  const float* Whh1 = (const float*)d_in[3];
  const float* bih1 = (const float*)d_in[4];
  const float* bhh1 = (const float*)d_in[5];
  const float* Wih2 = (const float*)d_in[6];
  const float* Whh2 = (const float*)d_in[7];
  const float* bih2 = (const float*)d_in[8];
  const float* bhh2 = (const float*)d_in[9];
  const float* Wih3 = (const float*)d_in[10];
  const float* Whh3 = (const float*)d_in[11];
  const float* bih3 = (const float*)d_in[12];
  const float* bhh3 = (const float*)d_in[13];
  const float* W1   = (const float*)d_in[14];
  const float* b1   = (const float*)d_in[15];
  const float* W2   = (const float*)d_in[16];
  const float* b2   = (const float*)d_in[17];
  const float* W3   = (const float*)d_in[18];
  const float* b3   = (const float*)d_in[19];
  const float* W4   = (const float*)d_in[20];
  const float* b4   = (const float*)d_in[21];

  _Float16* wsB = (_Float16*)d_ws;
  int Btot = in_sizes[0] / (T_STEPS * 2);     // x is [B, T, 2]

  prep_weights<<<NFRAG, 512, 0, stream>>>(Wih1, Whh1, Wih2, Whh2, Wih3, Whh3,
                                          W1, W2, W3, W4, wsB);
  lstm_fused<<<Btot / 128, 256, 0, stream>>>(x, y, bih1, bhh1, bih2, bhh2,
                                             bih3, bhh3, b1, b2, b3, b4,
                                             wsB, (float*)d_out);
}
